// NoiAware_50130858279661
// MI455X (gfx1250) — compile-verified, matrix operand order checked by reading, not run on hardware
//
#include <hip/hip_runtime.h>
#include <hip/hip_bf16.h>

typedef __attribute__((ext_vector_type(16))) _Float16 v16h;
typedef __attribute__((ext_vector_type(8)))  _Float16 v8h;
typedef __attribute__((ext_vector_type(4)))  _Float16 v4h;
typedef __attribute__((ext_vector_type(8)))  float    v8f;

#define B_SZ   4096
#define NEGS   64
#define EMB    128
#define HID    512
#define MARGINF 24.0f

// stable softplus(z) = log(1+exp(z))
__device__ __forceinline__ float softplusf(float z) {
    return fmaxf(z, 0.0f) + log1pf(expf(-fabsf(z)));
}

__device__ __forceinline__ float waveReduceSum32(float v) {
#pragma unroll
    for (int m = 16; m >= 1; m >>= 1) v += __shfl_xor(v, m, 32);
    return v;
}

// Kernel 1: positive triples -> f16 diff matrix (row-major B x 128) and
// pos_term[b] = -log_sigmoid(MARGIN - d_pos) = softplus(d_pos - MARGIN)
__global__ void diff_pos_kernel(const int* __restrict__ pos,
                                const float* __restrict__ ent,
                                const float* __restrict__ rel,
                                _Float16* __restrict__ diffh,
                                float* __restrict__ posterm) {
    int row  = blockIdx.x * 8 + (threadIdx.x >> 5);
    int lane = threadIdx.x & 31;
    if (row >= B_SZ) return;
    int h = pos[row * 3 + 0];
    int r = pos[row * 3 + 1];
    int t = pos[row * 3 + 2];
    float4 hv = ((const float4*)(ent + (size_t)h * EMB))[lane];
    float4 rv = ((const float4*)(rel + (size_t)r * EMB))[lane];
    float4 tv = ((const float4*)(ent + (size_t)t * EMB))[lane];
    float d0 = hv.x + rv.x - tv.x;
    float d1 = hv.y + rv.y - tv.y;
    float d2 = hv.z + rv.z - tv.z;
    float d3 = hv.w + rv.w - tv.w;
    v4h o;
    o[0] = (_Float16)d0; o[1] = (_Float16)d1; o[2] = (_Float16)d2; o[3] = (_Float16)d3;
    *(v4h*)(diffh + (size_t)row * EMB + lane * 4) = o;
    float s = fabsf(d0) + fabsf(d1) + fabsf(d2) + fabsf(d3);
    s = waveReduceSum32(s);
    if (lane == 0) posterm[row] = softplusf(s - MARGINF);
}

// Kernel 2: reorder W1 (EMB x HID f32, row-major) into f16 WMMA-B tile layout.
// Tile order: [ntile 0..31][kchunk 0..3][lane 0..31][slot 0..15] contiguous.
// Lane L supplies column n = ntile*16 + (L&15); base = (L<16)?0:8;
// slot s -> k = kc*32 + (s<8 ? base+s : 16+base+s-8)   (ISA 7.12.2 16-bit layout)
__global__ void w1_reorder_kernel(const float* __restrict__ W1,
                                  _Float16* __restrict__ W1t) {
    int idx  = blockIdx.x * 256 + threadIdx.x;   // 0..65535
    int slot = idx & 15;
    int lane = (idx >> 4) & 31;
    int kc   = (idx >> 9) & 3;
    int nt   = idx >> 11;
    int n    = nt * 16 + (lane & 15);
    int base = (lane < 16) ? 0 : 8;
    int k    = kc * 32 + ((slot < 8) ? (base + slot) : (16 + base + (slot - 8)));
    W1t[idx] = (_Float16)W1[k * HID + n];
}

// Kernel 3: negative gathers. One block per positive, 8 waves x 8 negatives.
__global__ void neg_kernel(const int* __restrict__ neg,
                           const float* __restrict__ ent,
                           const float* __restrict__ rel,
                           const int* __restrict__ nsamp,
                           float* __restrict__ negterm) {
    __shared__ float wpart[8];
    int b    = blockIdx.x;
    int wave = threadIdx.x >> 5;
    int lane = threadIdx.x & 31;
    float acc = 0.0f;
#pragma unroll 2
    for (int jj = 0; jj < 8; ++jj) {
        int j = wave * 8 + jj;
        size_t base = ((size_t)b * NEGS + j) * 3;
        int hi = neg[base + 0];
        int ri = neg[base + 1];
        int ti = neg[base + 2];
        float4 hv = ((const float4*)(ent + (size_t)hi * EMB))[lane];
        float4 rv = ((const float4*)(rel + (size_t)ri * EMB))[lane];
        float4 tv = ((const float4*)(ent + (size_t)ti * EMB))[lane];
        float s = fabsf(hv.x + rv.x - tv.x) + fabsf(hv.y + rv.y - tv.y) +
                  fabsf(hv.z + rv.z - tv.z) + fabsf(hv.w + rv.w - tv.w);
        s = waveReduceSum32(s);
        // log_sigmoid(MARGIN - d) = -softplus(d - MARGIN)
        acc += -softplusf(s - MARGINF);
    }
    if (lane == 0) wpart[wave] = acc;
    __syncthreads();
    if (threadIdx.x == 0) {
        float t = 0.0f;
#pragma unroll
        for (int w = 0; w < 8; ++w) t += wpart[w];
        negterm[b] = t / (float)nsamp[0];
    }
}

// Kernel 4: fused WMMA MLP: conf = sigmoid(relu(diff@W1 + b1) @ W2 + b2).
// One wave per 16-row M-tile; hid tile never leaves registers.
__global__ void mlp_wmma_kernel(const _Float16* __restrict__ diffh,
                                const _Float16* __restrict__ W1t,
                                const float* __restrict__ b1,
                                const float* __restrict__ W2,
                                const float* __restrict__ b2,
                                float* __restrict__ conf) {
    int wave  = threadIdx.x >> 5;
    int lane  = threadIdx.x & 31;
    int mtile = blockIdx.x * 4 + wave;          // 0..255
    int mrow  = lane & 15;
    int base  = (lane < 16) ? 0 : 8;

    // A operand: 4 K-chunks of the 16x128 diff tile in ISA 16-bit A layout.
    v16h A[4];
    size_t rowoff = (size_t)(mtile * 16 + mrow) * EMB;
#pragma unroll
    for (int kc = 0; kc < 4; ++kc) {
        v8h alo = *(const v8h*)(diffh + rowoff + kc * 32 + base);
        v8h ahi = *(const v8h*)(diffh + rowoff + kc * 32 + 16 + base);
#pragma unroll
        for (int i = 0; i < 8; ++i) { A[kc][i] = alo[i]; A[kc][i + 8] = ahi[i]; }
    }

    v8f acc = {};                               // relu(hid)*W2 partials per C slot
    for (int nt = 0; nt < 32; ++nt) {
        v8f c = {};
#pragma unroll
        for (int kc = 0; kc < 4; ++kc) {
            const v8h* bp = (const v8h*)(W1t + ((size_t)(nt * 4 + kc) * 32 + lane) * 16);
            v8h blo = bp[0], bhi = bp[1];
            v16h Bm;
#pragma unroll
            for (int i = 0; i < 8; ++i) { Bm[i] = blo[i]; Bm[i + 8] = bhi[i]; }
            c = __builtin_amdgcn_wmma_f32_16x16x32_f16(
                    false, A[kc], false, Bm, (short)0, c, false, false);
        }
        int col  = nt * 16 + (lane & 15);
        float bb = b1[col];
        float ww = W2[col];
#pragma unroll
        for (int r = 0; r < 8; ++r) {
            float hval = fmaxf(c[r] + bb, 0.0f);
            acc[r] += hval * ww;
        }
    }

    // C layout: lanes 0-15 hold rows r (N=lane), lanes 16-31 hold rows r+8.
    // Sum each VGPR slot across its 16-lane group.
#pragma unroll
    for (int r = 0; r < 8; ++r) {
        float v = acc[r];
#pragma unroll
        for (int m = 8; m >= 1; m >>= 1) v += __shfl_xor(v, m, 32);
        acc[r] = v;
    }
    if ((lane & 15) == 0) {
        float b2v = b2[0];
#pragma unroll
        for (int r = 0; r < 8; ++r) {
            int row = mtile * 16 + base + r;
            float logit = acc[r] + b2v;
            conf[row] = 1.0f / (1.0f + expf(-logit));
        }
    }
}

// Kernel 5: final scalar reduction.
__global__ void final_kernel(const float* __restrict__ conf,
                             const float* __restrict__ posterm,
                             const float* __restrict__ negterm,
                             float* __restrict__ out) {
    __shared__ float part[256];
    float s = 0.0f;
    for (int b = threadIdx.x; b < B_SZ; b += 256)
        s += conf[b] * (posterm[b] + negterm[b]);
    part[threadIdx.x] = s;
    __syncthreads();
    for (int off = 128; off > 0; off >>= 1) {
        if ((int)threadIdx.x < off) part[threadIdx.x] += part[threadIdx.x + off];
        __syncthreads();
    }
    if (threadIdx.x == 0) out[0] = part[0];
}

extern "C" void kernel_launch(void* const* d_in, const int* in_sizes, int n_in,
                              void* d_out, int out_size, void* d_ws, size_t ws_size,
                              hipStream_t stream) {
    (void)in_sizes; (void)n_in; (void)out_size; (void)ws_size;
    const int*   pos   = (const int*)d_in[0];
    const int*   neg   = (const int*)d_in[1];
    const int*   nsamp = (const int*)d_in[2];
    const float* ent   = (const float*)d_in[3];
    const float* rel   = (const float*)d_in[4];
    const float* W1    = (const float*)d_in[5];
    const float* b1    = (const float*)d_in[6];
    const float* W2    = (const float*)d_in[7];
    const float* b2    = (const float*)d_in[8];

    char* ws = (char*)d_ws;
    _Float16* diffh   = (_Float16*)(ws + 0);            // 4096*128*2 = 1,048,576 B
    _Float16* W1t     = (_Float16*)(ws + 1048576);      // 128*512*2  =   131,072 B
    float*    posterm = (float*)   (ws + 1179648);      // 4096*4
    float*    negterm = (float*)   (ws + 1196032);      // 4096*4
    float*    conf    = (float*)   (ws + 1212416);      // 4096*4

    diff_pos_kernel <<<512,  256, 0, stream>>>(pos, ent, rel, diffh, posterm);
    w1_reorder_kernel<<<256, 256, 0, stream>>>(W1, W1t);
    neg_kernel      <<<4096, 256, 0, stream>>>(neg, ent, rel, nsamp, negterm);
    mlp_wmma_kernel <<<64,   128, 0, stream>>>(diffh, W1t, b1, W2, b2, conf);
    final_kernel    <<<1,    256, 0, stream>>>(conf, posterm, negterm, (float*)d_out);
}
